// RelationalGraphConv_9577777070223
// MI455X (gfx1250) — matrix operand order repaired and checked by reading, not compiled, not run on hardware
//
#include <hip/hip_runtime.h>
#include <hip/hip_bf16.h>

#define N_NODES 50000
#define N_RELS 8
#define N_EDGES 100000
#define DIM 128
#define NUM_BASES 8
#define TILES_PER_REL (N_EDGES / 16)      // 6250
#define ROW_TILES (N_NODES / 16)          // 3125 (50000 = 16*3125 exactly)
#define WP_ELEMS (9 * DIM * DIM)          // 9 matrices (8 rel + self), bf16
#define REL_SLOT (4 * 8 * 32 * 16)        // 16384 bf16 = 32KB per matrix slot

typedef __attribute__((ext_vector_type(16))) __bf16 v16bf;
typedef __attribute__((ext_vector_type(8)))  __bf16 v8bf;
typedef __attribute__((ext_vector_type(8)))  float  v8f;

__device__ __forceinline__ __bf16 f2bf(float f) {
  unsigned int u = __builtin_bit_cast(unsigned int, f);
  u += 0x7FFFu + ((u >> 16) & 1u);               // round-to-nearest-even
  unsigned short h = (unsigned short)(u >> 16);
  return __builtin_bit_cast(__bf16, h);
}

// ---------------- weight basis combine + bf16 pack into WMMA B-frag layout ----
// B-frag (32x16, bf16): lane<16 -> col N=lane, K={0..7,16..23} ; lane>=16 -> col N=lane-16, K={8..15,24..31}
// wp index: (((r*4 + kc)*8 + nt)*32 + lane)*16 + t   (t = element slot within lane)
__global__ void pack_weights_kernel(const float* __restrict__ bases,
                                    const float* __restrict__ coeff,
                                    const float* __restrict__ selfw,
                                    __bf16* __restrict__ wp) {
  int gid = blockIdx.x * blockDim.x + threadIdx.x;
  if (gid >= WP_ELEMS) return;
  int r = gid / (DIM * DIM);
  int rem = gid - r * DIM * DIM;
  int k = rem / DIM;
  int o = rem - k * DIM;
  float w;
  if (r < N_RELS) {
    w = 0.f;
#pragma unroll
    for (int b = 0; b < NUM_BASES; ++b)
      w += coeff[r * NUM_BASES + b] * bases[(b * DIM + k) * DIM + o];
  } else {
    w = selfw[k * DIM + o];
  }
  int kc   = k >> 5;
  int kin  = k & 31;
  int lane = (((kin >> 3) & 1) << 4) + (o & 15);
  int t    = (kin & 7) + ((kin >> 4) << 3);
  int nt   = o >> 4;
  wp[((((r * 4 + kc) * 8 + nt) * 32 + lane) << 4) + t] = f2bf(w);
}

// ---------------- feat f32 -> bf16 copy (one-shot, removes conv from hot loop)
__global__ void feat_bf16_kernel(const float* __restrict__ feat, __bf16* __restrict__ fb) {
  int gid = blockIdx.x * blockDim.x + threadIdx.x;
  if (gid < N_NODES * DIM) fb[gid] = f2bf(feat[gid]);
}

// ---------------- count / invert kernels ------------------------------------
__global__ void zero_cnt_kernel(float* __restrict__ cnt) {
  int gid = blockIdx.x * blockDim.x + threadIdx.x;
  if (gid < N_RELS * N_NODES) cnt[gid] = 0.f;
}

__global__ void count_kernel(const int* __restrict__ edge_dst, float* __restrict__ cnt) {
  int gid = blockIdx.x * blockDim.x + threadIdx.x;
  if (gid >= N_RELS * N_EDGES) return;
  int r = gid / N_EDGES;
  atomicAdd(&cnt[r * N_NODES + edge_dst[gid]], 1.0f);
}

__global__ void recip_kernel(float* __restrict__ cnt) {
  int gid = blockIdx.x * blockDim.x + threadIdx.x;
  if (gid < N_RELS * N_NODES) cnt[gid] = 1.0f / fmaxf(cnt[gid], 1.0f);
}

// ---------------- A-fragment gather from bf16 feat (16x32 bf16 ISA layout) ---
// lane<16: row M=lane, K = kc*32 + {0..7, 16..23}
// lane>=16: row M=lane-16, K = kc*32 + {8..15, 24..31}
__device__ __forceinline__ v16bf load_a_frag(const __bf16* __restrict__ fb, int kc, int hh) {
  int b1 = (kc << 5) + (hh << 3);
  v8bf lo = *reinterpret_cast<const v8bf*>(fb + b1);
  v8bf hi = *reinterpret_cast<const v8bf*>(fb + b1 + 16);
  v16bf a;
#pragma unroll
  for (int t = 0; t < 8; ++t) { a[t] = lo[t]; a[t + 8] = hi[t]; }
  return a;
}

#define WMMA_BF16(A, B, C) \
  __builtin_amdgcn_wmma_f32_16x16x32_bf16(false, (A), false, (B), (short)0, (C), false, false)

__device__ __forceinline__ v16bf lds_b_frag(const __bf16* lw, int kc, int nt, int lane) {
  return *reinterpret_cast<const v16bf*>(lw + (((kc * 8 + nt) * 32 + lane) << 4));
}

// ---------------- main: per-relation gathered GEMM + scaled atomic scatter ---
// blockIdx.y = relation; 32KB of that relation's weights staged in LDS once per
// block and shared by all 8 waves (1 wave = one 16-edge tile).
// out[dst] += invcnt[r][dst] * (feat[src] @ W_r)
__global__ void __launch_bounds__(256) rel_scatter_gemm_kernel(
    const __bf16* __restrict__ fb,
    const int* __restrict__ edge_src,
    const int* __restrict__ edge_dst,
    const __bf16* __restrict__ wp,
    const float* __restrict__ invcnt,
    float* __restrict__ out) {
  __shared__ __attribute__((aligned(32))) __bf16 lw[REL_SLOT];
  int rel = blockIdx.y;

  // cooperative stage: 32KB = 2048 uint4, 8 per thread
  {
    const uint4* gsrc = reinterpret_cast<const uint4*>(wp + (size_t)rel * REL_SLOT);
    uint4* ldst = reinterpret_cast<uint4*>(lw);
#pragma unroll
    for (int i = 0; i < 8; ++i) ldst[threadIdx.x + 256 * i] = gsrc[threadIdx.x + 256 * i];
  }
  __syncthreads();

  int wave = threadIdx.x >> 5;
  int lane = threadIdx.x & 31;
  int tile = blockIdx.x * 8 + wave;
  if (tile >= TILES_PER_REL) return;           // wave-uniform: EXEC stays all-1s
  int e0 = tile << 4;
  int hh = lane >> 4;
  int m  = lane & 15;

  int srow = edge_src[rel * N_EDGES + e0 + m];
  const __bf16* fr = fb + srow * DIM;
  v16bf a0 = load_a_frag(fr, 0, hh);
  v16bf a1 = load_a_frag(fr, 1, hh);
  v16bf a2 = load_a_frag(fr, 2, hh);
  v16bf a3 = load_a_frag(fr, 3, hh);

  // rows this lane scatters: M = hh*8 + j  (D-matrix VGPR layout)
  int   obase[8];
  float scl[8];
#pragma unroll
  for (int j = 0; j < 8; ++j) {
    int d    = edge_dst[rel * N_EDGES + e0 + (hh << 3) + j];
    obase[j] = d * DIM + m;                    // 32-bit offset -> saddr atomics
    scl[j]   = invcnt[rel * N_NODES + d];
  }

#pragma unroll
  for (int nt = 0; nt < 8; ++nt) {
    v8f acc = {};
    acc = WMMA_BF16(a0, lds_b_frag(lw, 0, nt, lane), acc);
    acc = WMMA_BF16(a1, lds_b_frag(lw, 1, nt, lane), acc);
    acc = WMMA_BF16(a2, lds_b_frag(lw, 2, nt, lane), acc);
    acc = WMMA_BF16(a3, lds_b_frag(lw, 3, nt, lane), acc);
#pragma unroll
    for (int j = 0; j < 8; ++j)
      atomicAdd(out + obase[j] + (nt << 4), acc[j] * scl[j]);
  }
}

// ---------------- self loop: out = feat @ W_self (initializes d_out) ---------
__global__ void __launch_bounds__(256) self_gemm_kernel(
    const __bf16* __restrict__ fb,
    const __bf16* __restrict__ wp,
    float* __restrict__ out) {
  __shared__ __attribute__((aligned(32))) __bf16 lw[REL_SLOT];
  {
    const uint4* gsrc = reinterpret_cast<const uint4*>(wp + (size_t)8 * REL_SLOT);
    uint4* ldst = reinterpret_cast<uint4*>(lw);
#pragma unroll
    for (int i = 0; i < 8; ++i) ldst[threadIdx.x + 256 * i] = gsrc[threadIdx.x + 256 * i];
  }
  __syncthreads();

  int wave = threadIdx.x >> 5;
  int lane = threadIdx.x & 31;
  int tile = blockIdx.x * 8 + wave;
  if (tile >= ROW_TILES) return;               // wave-uniform
  int hh = lane >> 4;
  int m  = lane & 15;
  int row = (tile << 4) + m;
  const __bf16* fr = fb + row * DIM;
  v16bf a0 = load_a_frag(fr, 0, hh);
  v16bf a1 = load_a_frag(fr, 1, hh);
  v16bf a2 = load_a_frag(fr, 2, hh);
  v16bf a3 = load_a_frag(fr, 3, hh);

  int obase = ((tile << 4) + (hh << 3)) * DIM + m;   // row M=hh*8+j stored at obase+j*DIM
#pragma unroll
  for (int nt = 0; nt < 8; ++nt) {
    v8f acc = {};
    acc = WMMA_BF16(a0, lds_b_frag(lw, 0, nt, lane), acc);
    acc = WMMA_BF16(a1, lds_b_frag(lw, 1, nt, lane), acc);
    acc = WMMA_BF16(a2, lds_b_frag(lw, 2, nt, lane), acc);
    acc = WMMA_BF16(a3, lds_b_frag(lw, 3, nt, lane), acc);
#pragma unroll
    for (int j = 0; j < 8; ++j)
      out[obase + j * DIM + (nt << 4)] = acc[j];
  }
}

// ---------------- in-place LayerNorm (one wave per 128-wide row) -------------
__global__ void __launch_bounds__(256) layernorm_kernel(
    float* __restrict__ out,
    const float* __restrict__ gamma,
    const float* __restrict__ beta) {
  int wave = threadIdx.x >> 5;
  int lane = threadIdx.x & 31;
  int row  = blockIdx.x * 8 + wave;            // 6250 blocks * 8 = 50000 rows
  float* rp = out + (size_t)row * DIM + lane * 4;
  float4 v = *reinterpret_cast<const float4*>(rp);
  float s = v.x + v.y + v.z + v.w;
#pragma unroll
  for (int k = 16; k >= 1; k >>= 1) s += __shfl_xor(s, k, 32);
  float mean = s * (1.0f / 128.0f);
  float dx = v.x - mean, dy = v.y - mean, dz = v.z - mean, dw = v.w - mean;
  float q = dx * dx + dy * dy + dz * dz + dw * dw;
#pragma unroll
  for (int k = 16; k >= 1; k >>= 1) q += __shfl_xor(q, k, 32);
  float rs = rsqrtf(q * (1.0f / 128.0f) + 1e-5f);
  float4 g  = *reinterpret_cast<const float4*>(gamma + lane * 4);
  float4 bt = *reinterpret_cast<const float4*>(beta + lane * 4);
  float4 r;
  r.x = dx * rs * g.x + bt.x;
  r.y = dy * rs * g.y + bt.y;
  r.z = dz * rs * g.z + bt.z;
  r.w = dw * rs * g.w + bt.w;
  *reinterpret_cast<float4*>(rp) = r;
}

extern "C" void kernel_launch(void* const* d_in, const int* in_sizes, int n_in,
                              void* d_out, int out_size, void* d_ws, size_t ws_size,
                              hipStream_t stream) {
  const float* feat     = (const float*)d_in[0];
  const int*   edge_src = (const int*)d_in[1];
  const int*   edge_dst = (const int*)d_in[2];
  const float* bases    = (const float*)d_in[3];
  const float* coeff    = (const float*)d_in[4];
  const float* selfw    = (const float*)d_in[5];
  const float* gamma    = (const float*)d_in[6];
  const float* beta     = (const float*)d_in[7];
  float* out = (float*)d_out;

  // ws layout: [0,288KB) packed bf16 weights | [8*N] f32 inv-counts | bf16 feat
  __bf16* wp    = (__bf16*)d_ws;
  float* invcnt = (float*)((char*)d_ws + (size_t)WP_ELEMS * sizeof(__bf16));
  __bf16* fb    = (__bf16*)((char*)invcnt + (size_t)N_RELS * N_NODES * sizeof(float));

  zero_cnt_kernel<<<(N_RELS * N_NODES + 255) / 256, 256, 0, stream>>>(invcnt);
  pack_weights_kernel<<<(WP_ELEMS + 255) / 256, 256, 0, stream>>>(bases, coeff, selfw, wp);
  feat_bf16_kernel<<<(N_NODES * DIM + 255) / 256, 256, 0, stream>>>(feat, fb);
  count_kernel<<<(N_RELS * N_EDGES + 255) / 256, 256, 0, stream>>>(edge_dst, invcnt);
  recip_kernel<<<(N_RELS * N_NODES + 255) / 256, 256, 0, stream>>>(invcnt);
  self_gemm_kernel<<<(ROW_TILES + 7) / 8, 256, 0, stream>>>(fb, wp, out);
  dim3 sgrid((TILES_PER_REL + 7) / 8, N_RELS);
  rel_scatter_gemm_kernel<<<sgrid, 256, 0, stream>>>(fb, edge_src, edge_dst,
                                                     wp, invcnt, out);
  layernorm_kernel<<<N_NODES / 8, 256, 0, stream>>>(out, gamma, beta);
}
